// Block_40879498729310
// MI455X (gfx1250) — compile-verified, matrix operand order checked by reading, not run on hardware
//
#include <hip/hip_runtime.h>
#include <hip/hip_bf16.h>
#include <math.h>

// ---------------------------------------------------------------------------
// GPT-2 block for MI455X (gfx1250, wave32, WMMA).
// All GEMMs + attention matmuls use v_wmma_f32_16x16x32_f16 (f32 accumulate).
// GEMM tiles are staged into double-buffered LDS with the CDNA5 async path
// (GLOBAL_LOAD_ASYNC_TO_LDS_B128 + s_wait_asynccnt); weights are converted
// f32->f16 and pre-transposed once per launch; the QKV GEMM writes V directly
// in [b][h][d][t] layout so flash-attention V fragments are contiguous 16B.
// ---------------------------------------------------------------------------

typedef __attribute__((ext_vector_type(16))) _Float16 v16h;
typedef __attribute__((ext_vector_type(8)))  _Float16 h8;
typedef __attribute__((ext_vector_type(8)))  float    v8f;

// exact pointee type of __builtin_amdgcn_global_load_async_to_lds_b128 params
// (from compiler diagnostic: int vector_size(16), AS1 src / AS3 dst)
typedef int v4i __attribute__((vector_size(16)));
typedef __attribute__((address_space(1))) v4i as1_v4i;
typedef __attribute__((address_space(3))) v4i as3_v4i;

#define D_EMBD 768
#define N_HEAD 12
#define HD     64
#define SEQ_T  2048
#define ROWS   4096   // B*T

#if defined(__has_builtin)
#  if __has_builtin(__builtin_amdgcn_global_load_async_to_lds_b128) && \
      __has_builtin(__builtin_amdgcn_s_wait_asynccnt)
#    define USE_ASYNC_LDS 1
#  else
#    define USE_ASYNC_LDS 0
#  endif
#else
#  define USE_ASYNC_LDS 0
#endif

static __device__ __forceinline__ v8f wmma16x16x32(v16h a, v16h b, v8f c) {
  // 8 args: (neg_a, A, neg_b, B, c_mod, C, reuse_a, reuse_b)
  return __builtin_amdgcn_wmma_f32_16x16x32_f16(false, a, false, b, (short)0, c,
                                                false, false);
}

static __device__ __forceinline__ float gelu_tanh(float v) {
  return 0.5f * v * (1.0f + tanhf(0.7978845608028654f * (v + 0.044715f * v * v * v)));
}

// ---------------------------------------------------------------------------
// f32 [K][N] -> f16 [N][K] tiled transpose (weights, once per launch).
// ---------------------------------------------------------------------------
__global__ __launch_bounds__(256)
void cvt_transpose(const float* __restrict__ in, _Float16* __restrict__ out,
                   int K, int N) {
  __shared__ float tile[32][33];
  const int n0 = blockIdx.x * 32, k0 = blockIdx.y * 32;
  const int tx = threadIdx.x & 31, ty = threadIdx.x >> 5;
#pragma unroll
  for (int i = 0; i < 32; i += 8)
    tile[ty + i][tx] = in[(size_t)(k0 + ty + i) * N + n0 + tx];
  __syncthreads();
#pragma unroll
  for (int i = 0; i < 32; i += 8)
    out[(size_t)(n0 + ty + i) * K + k0 + tx] = (_Float16)tile[tx][ty + i];
}

// ---------------------------------------------------------------------------
// LayerNorm: one row (768 cols) per 256-thread block, f32 stats, f16 output.
// ---------------------------------------------------------------------------
__global__ __launch_bounds__(256)
void ln_kernel(const float* __restrict__ x, const float* __restrict__ w,
               const float* __restrict__ b, _Float16* __restrict__ out, int cols) {
  const int row = blockIdx.x;
  const float* xr = x + (size_t)row * cols;
  float s = 0.f, q = 0.f;
  for (int i = threadIdx.x; i < cols; i += 256) {
    float v = xr[i];
    s += v; q += v * v;
  }
#pragma unroll
  for (int m = 16; m >= 1; m >>= 1) {
    s += __shfl_xor(s, m, 32);
    q += __shfl_xor(q, m, 32);
  }
  __shared__ float ss[8], sq[8];
  __shared__ float stat[2];
  const int wv = threadIdx.x >> 5, lane = threadIdx.x & 31;
  if (lane == 0) { ss[wv] = s; sq[wv] = q; }
  __syncthreads();
  if (threadIdx.x == 0) {
    float ts = 0.f, tq = 0.f;
#pragma unroll
    for (int i = 0; i < 8; ++i) { ts += ss[i]; tq += sq[i]; }
    float mu = ts / (float)cols;
    float var = tq / (float)cols - mu * mu;
    stat[0] = mu;
    stat[1] = rsqrtf(var + 1e-5f);
  }
  __syncthreads();
  const float mu = stat[0], rs = stat[1];
  for (int i = threadIdx.x; i < cols; i += 256)
    out[(size_t)row * cols + i] = (_Float16)((xr[i] - mu) * rs * w[i] + b[i]);
}

// ---------------------------------------------------------------------------
// WMMA GEMM: C[M,N] = A[M,K](f16) @ Wt[N,K](f16, pre-transposed) + bias,
// optional GELU, optional f32 residual add; f32 and/or f16 output.
// If vT != nullptr (QKV GEMM), columns [1536,2304) are stored transposed as
// vT[b][h][d][t] with one contiguous 16B store per fragment per lane.
// Block: 256 threads (8 waves), tile 128x128, BK=32; wave -> 64x32 patch.
// Double-buffered LDS staging overlapped with the WMMA chain.
// ---------------------------------------------------------------------------
__global__ __launch_bounds__(256)
void gemm_kernel(const _Float16* __restrict__ A, const _Float16* __restrict__ Wt,
                 const float* __restrict__ bias, int M, int N, int K,
                 float* __restrict__ outF, _Float16* __restrict__ outH,
                 const float* __restrict__ resid, int do_gelu,
                 _Float16* __restrict__ vT) {
  __shared__ _Float16 lA[2][128 * 32];
  __shared__ _Float16 lB[2][128 * 32];

  const int tid  = threadIdx.x;
  const int lane = tid & 31;
  const int wv   = tid >> 5;
  const int wm   = wv >> 2;           // 0..1
  const int wn   = wv & 3;            // 0..3
  const int bm   = blockIdx.y;
  const int bn   = blockIdx.x;
  const int row16 = lane & 15;
  const int hi8   = (lane >> 4) << 3; // 0 or 8: wave32 half selects K sub-offset

  const int sr = tid >> 1, sc = (tid & 1) * 16;   // staging row / col-chunk
  const int sOff = sr * 32 + sc;
  const _Float16* gA = A  + (size_t)(bm * 128 + sr) * K + sc;
  const _Float16* gB = Wt + (size_t)(bn * 128 + sr) * K + sc;

  v8f acc[4][2];
  const v8f vz = {};
#pragma unroll
  for (int i = 0; i < 4; ++i)
#pragma unroll
    for (int j = 0; j < 2; ++j) acc[i][j] = vz;

  // fragment compute on buffer `buf`
  auto compute = [&](int buf) {
    v16h af[4], bf[2];
#pragma unroll
    for (int mr = 0; mr < 4; ++mr) {
      const _Float16* p = &lA[buf][(wm * 64 + mr * 16 + row16) * 32 + hi8];
      h8 lo = *(const h8*)p;
      h8 hi = *(const h8*)(p + 16);
#pragma unroll
      for (int i = 0; i < 8; ++i) { af[mr][i] = lo[i]; af[mr][i + 8] = hi[i]; }
    }
#pragma unroll
    for (int nc = 0; nc < 2; ++nc) {
      const _Float16* p = &lB[buf][(wn * 32 + nc * 16 + row16) * 32 + hi8];
      h8 lo = *(const h8*)p;
      h8 hi = *(const h8*)(p + 16);
#pragma unroll
      for (int i = 0; i < 8; ++i) { bf[nc][i] = lo[i]; bf[nc][i + 8] = hi[i]; }
    }
#pragma unroll
    for (int mr = 0; mr < 4; ++mr)
#pragma unroll
      for (int nc = 0; nc < 2; ++nc)
        acc[mr][nc] = wmma16x16x32(af[mr], bf[nc], acc[mr][nc]);
  };

#if USE_ASYNC_LDS
  // CDNA5 async global->LDS staging (no VGPR round-trip, ASYNCcnt tracked)
  auto stage = [&](int k0, int buf) {
    __builtin_amdgcn_global_load_async_to_lds_b128(
        (as1_v4i*)(gA + k0),     (as3_v4i*)&lA[buf][sOff],     0, 0);
    __builtin_amdgcn_global_load_async_to_lds_b128(
        (as1_v4i*)(gA + k0 + 8), (as3_v4i*)&lA[buf][sOff + 8], 0, 0);
    __builtin_amdgcn_global_load_async_to_lds_b128(
        (as1_v4i*)(gB + k0),     (as3_v4i*)&lB[buf][sOff],     0, 0);
    __builtin_amdgcn_global_load_async_to_lds_b128(
        (as1_v4i*)(gB + k0 + 8), (as3_v4i*)&lB[buf][sOff + 8], 0, 0);
  };
  stage(0, 0);
  __builtin_amdgcn_s_wait_asynccnt(0);
  __syncthreads();
  int cur = 0;
  for (int k0 = 0; k0 < K; k0 += 32, cur ^= 1) {
    const bool more = (k0 + 32 < K);
    if (more) {
      stage(k0 + 32, cur ^ 1);                 // overlap with WMMA below
      if (k0 + 64 < K) {
        __builtin_prefetch(gA + k0 + 64, 0, 1);
        __builtin_prefetch(gB + k0 + 64, 0, 1);
      }
    }
    compute(cur);
    if (more) __builtin_amdgcn_s_wait_asynccnt(0);
    __syncthreads();
  }
#else
  // software-pipelined sync staging: loads issued before compute, LDS stores
  // after, so global latency hides behind the WMMA chain.
  {
    h8 a0 = *(const h8*)gA, a1 = *(const h8*)(gA + 8);
    h8 b0 = *(const h8*)gB, b1 = *(const h8*)(gB + 8);
    *(h8*)&lA[0][sOff] = a0;  *(h8*)&lA[0][sOff + 8] = a1;
    *(h8*)&lB[0][sOff] = b0;  *(h8*)&lB[0][sOff + 8] = b1;
  }
  __syncthreads();
  int cur = 0;
  for (int k0 = 0; k0 < K; k0 += 32, cur ^= 1) {
    const bool more = (k0 + 32 < K);
    h8 a0, a1, b0, b1;
    if (more) {
      a0 = *(const h8*)(gA + k0 + 32);  a1 = *(const h8*)(gA + k0 + 40);
      b0 = *(const h8*)(gB + k0 + 32);  b1 = *(const h8*)(gB + k0 + 40);
      if (k0 + 64 < K) {
        __builtin_prefetch(gA + k0 + 64, 0, 1);
        __builtin_prefetch(gB + k0 + 64, 0, 1);
      }
    }
    compute(cur);
    if (more) {
      *(h8*)&lA[cur ^ 1][sOff] = a0;  *(h8*)&lA[cur ^ 1][sOff + 8] = a1;
      *(h8*)&lB[cur ^ 1][sOff] = b0;  *(h8*)&lB[cur ^ 1][sOff + 8] = b1;
    }
    __syncthreads();
  }
#endif

  // --- epilogue: C layout = VGPR r -> row r (lanes 0-15) / r+8 (lanes 16-31)
#pragma unroll
  for (int mr = 0; mr < 4; ++mr)
#pragma unroll
    for (int nc = 0; nc < 2; ++nc) {
      const int gcol  = bn * 128 + wn * 32 + nc * 16 + row16;
      const int rbase = bm * 128 + wm * 64 + mr * 16 + hi8;   // + r, r=0..7
      if (vT && gcol >= 2 * D_EMBD) {
        // V column: store transposed vT[b][h][d][t], 8 consecutive t = 16B
        const int dglob = gcol - 2 * D_EMBD;
        const int hh = dglob >> 6, dl = dglob & 63;
        const int bb = rbase >> 11, t0 = rbase & 2047;
        h8 pk;
#pragma unroll
        for (int r = 0; r < 8; ++r)
          pk[r] = (_Float16)(acc[mr][nc][r] + bias[gcol]);
        *(h8*)&vT[(((size_t)bb * N_HEAD + hh) * HD + dl) * SEQ_T + t0] = pk;
      } else {
#pragma unroll
        for (int r = 0; r < 8; ++r) {
          const int grow = rbase + r;
          float v = acc[mr][nc][r] + bias[gcol];
          if (do_gelu) v = gelu_tanh(v);
          size_t idx = (size_t)grow * N + gcol;
          if (resid) v += resid[idx];
          if (outF) outF[idx] = v;
          if (outH) outH[idx] = (_Float16)v;
        }
      }
    }
}

// ---------------------------------------------------------------------------
// Flash attention, causal. Grid: B * N_HEAD * (T/64) blocks, 128 threads.
// Each of 4 waves owns 16 query rows; online softmax over 32-wide key blocks.
// ---------------------------------------------------------------------------
__global__ __launch_bounds__(128)
void attn_kernel(const _Float16* __restrict__ qkv, const _Float16* __restrict__ vT,
                 _Float16* __restrict__ y) {
  __shared__ _Float16 ldsP[4][16][32];

  const int lane  = threadIdx.x & 31;
  const int wv    = threadIdx.x >> 5;
  const int qt    = blockIdx.x & 31;              // 32 q-tiles of 64 rows
  const int h     = (blockIdx.x >> 5) % N_HEAD;
  const int bb    = blockIdx.x / (32 * N_HEAD);
  const int q0    = qt * 64 + wv * 16;
  const int row16 = lane & 15;
  const int hi8   = (lane >> 4) << 3;

  // Q fragments (A layout), pre-scaled by 1/sqrt(hd) = 0.125
  v16h qf[2];
  {
    const _Float16* qbase =
        qkv + ((size_t)(bb * SEQ_T + q0 + row16)) * (3 * D_EMBD) + h * HD;
#pragma unroll
    for (int dc = 0; dc < 2; ++dc) {
      int dlo = dc * 32 + hi8;
      h8 lo = *(const h8*)(qbase + dlo);
      h8 hi = *(const h8*)(qbase + dlo + 16);
#pragma unroll
      for (int i = 0; i < 8; ++i) {
        qf[dc][i]     = (_Float16)((float)lo[i] * 0.125f);
        qf[dc][i + 8] = (_Float16)((float)hi[i] * 0.125f);
      }
    }
  }

  float m[8], l[8], alpha[8];
  v8f o[4];
  const v8f vz = {};
#pragma unroll
  for (int r = 0; r < 8; ++r) { m[r] = -INFINITY; l[r] = 0.f; }
#pragma unroll
  for (int f = 0; f < 4; ++f) o[f] = vz;

  const int kblocks = 2 * qt + 2;   // uniform across the block (causal bound)
  for (int kb = 0; kb < kblocks; ++kb) {
    const int k0 = kb * 32;

    // K^T fragments (B layout, K-dim = d): two col tiles x two d chunks
    v16h kf[2][2];
#pragma unroll
    for (int ct = 0; ct < 2; ++ct) {
      const _Float16* kbase =
          qkv + ((size_t)(bb * SEQ_T + k0 + ct * 16 + row16)) * (3 * D_EMBD) +
          D_EMBD + h * HD;
#pragma unroll
      for (int dc = 0; dc < 2; ++dc) {
        int dlo = dc * 32 + hi8;
        h8 lo = *(const h8*)(kbase + dlo);
        h8 hi = *(const h8*)(kbase + dlo + 16);
#pragma unroll
        for (int i = 0; i < 8; ++i) { kf[ct][dc][i] = lo[i]; kf[ct][dc][i + 8] = hi[i]; }
      }
    }

    v8f s[2];
#pragma unroll
    for (int ct = 0; ct < 2; ++ct) {
      s[ct] = wmma16x16x32(qf[0], kf[ct][0], vz);
      s[ct] = wmma16x16x32(qf[1], kf[ct][1], s[ct]);
    }

    // online softmax (row = q0 + r + hi8; col = k0 + ct*16 + row16)
    float pv[2][8];
#pragma unroll
    for (int r = 0; r < 8; ++r) {
      const int qrow = q0 + r + hi8;
      float sv[2];
      float bmax = -INFINITY;
#pragma unroll
      for (int ct = 0; ct < 2; ++ct) {
        int kcol = k0 + ct * 16 + row16;
        float v = s[ct][r];
        if (kcol > qrow) v = -INFINITY;
        sv[ct] = v;
        bmax = fmaxf(bmax, v);
      }
#pragma unroll
      for (int msk = 8; msk >= 1; msk >>= 1)
        bmax = fmaxf(bmax, __shfl_xor(bmax, msk, 32));
      float nm = fmaxf(m[r], bmax);
      float a  = expf(m[r] - nm);          // 0 on first block (m=-inf, nm finite)
      float ps = 0.f;
#pragma unroll
      for (int ct = 0; ct < 2; ++ct) {
        float p = expf(sv[ct] - nm);
        pv[ct][r] = p;
        ps += p;
      }
#pragma unroll
      for (int msk = 8; msk >= 1; msk >>= 1) ps += __shfl_xor(ps, msk, 32);
      l[r] = l[r] * a + ps;
      m[r] = nm;
      alpha[r] = a;
    }
#pragma unroll
    for (int f = 0; f < 4; ++f)
#pragma unroll
      for (int r = 0; r < 8; ++r) o[f][r] *= alpha[r];

    // D-frag -> A-frag conversion through LDS (per-wave private region)
    __syncthreads();
#pragma unroll
    for (int r = 0; r < 8; ++r)
#pragma unroll
      for (int ct = 0; ct < 2; ++ct)
        ldsP[wv][r + hi8][ct * 16 + row16] = (_Float16)pv[ct][r];
    __syncthreads();

    v16h pf;
    {
      const _Float16* pb = &ldsP[wv][row16][hi8];
      h8 lo = *(const h8*)pb;
      h8 hi = *(const h8*)(pb + 16);
#pragma unroll
      for (int i = 0; i < 8; ++i) { pf[i] = lo[i]; pf[i + 8] = hi[i]; }
    }

    // V fragments (B layout, K-dim = key index) from vT[b][h][d][t]:
    // contiguous over t -> two 16B loads per fragment
#pragma unroll
    for (int dt = 0; dt < 4; ++dt) {
      v16h vf;
      const _Float16* vb =
          vT + (((size_t)bb * N_HEAD + h) * HD + dt * 16 + row16) * SEQ_T + k0;
      h8 lo = *(const h8*)(vb + hi8);
      h8 hi = *(const h8*)(vb + hi8 + 16);
#pragma unroll
      for (int i = 0; i < 8; ++i) { vf[i] = lo[i]; vf[i + 8] = hi[i]; }
      o[dt] = wmma16x16x32(pf, vf, o[dt]);
    }
  }

  // normalize + write y[b, q, h, d] as f16
#pragma unroll
  for (int dt = 0; dt < 4; ++dt)
#pragma unroll
    for (int r = 0; r < 8; ++r) {
      int qrow = q0 + r + hi8;
      int d    = dt * 16 + row16;
      y[((size_t)(bb * SEQ_T + qrow)) * D_EMBD + h * HD + d] =
          (_Float16)(o[dt][r] / l[r]);
    }
}

// ---------------------------------------------------------------------------
// Host launcher
// ---------------------------------------------------------------------------
extern "C" void kernel_launch(void* const* d_in, const int* in_sizes, int n_in,
                              void* d_out, int out_size, void* d_ws, size_t ws_size,
                              hipStream_t stream) {
  (void)in_sizes; (void)n_in; (void)out_size; (void)ws_size;
  const float* x      = (const float*)d_in[0];
  const float* ln1_w  = (const float*)d_in[1];
  const float* ln1_b  = (const float*)d_in[2];
  const float* attn_w = (const float*)d_in[3];
  const float* attn_b = (const float*)d_in[4];
  const float* proj_w = (const float*)d_in[5];
  const float* proj_b = (const float*)d_in[6];
  const float* ln2_w  = (const float*)d_in[7];
  const float* ln2_b  = (const float*)d_in[8];
  const float* fc_w   = (const float*)d_in[9];
  const float* fc_b   = (const float*)d_in[10];
  const float* fc2_w  = (const float*)d_in[11];
  const float* fc2_b  = (const float*)d_in[12];

  char* ws = (char*)d_ws;
  size_t off = 0;
  auto take = [&](size_t bytes) -> char* {
    char* p = ws + off;
    off += (bytes + 255) & ~(size_t)255;
    return p;
  };
  // weights: f16, transposed [N][K]
  _Float16* wAt  = (_Float16*)take((size_t)2304 * 768 * 2);
  _Float16* wPt  = (_Float16*)take((size_t)768 * 768 * 2);
  _Float16* wFt  = (_Float16*)take((size_t)3072 * 768 * 2);
  _Float16* wF2t = (_Float16*)take((size_t)768 * 3072 * 2);
  // activations
  _Float16* xn   = (_Float16*)take((size_t)ROWS * 768 * 2);
  _Float16* qkvh = (_Float16*)take((size_t)ROWS * 2304 * 2);  // V region unused
  _Float16* vT   = (_Float16*)take((size_t)2 * N_HEAD * HD * SEQ_T * 2);
  _Float16* yh   = (_Float16*)take((size_t)ROWS * 768 * 2);
  float*    x2   = (float*)   take((size_t)ROWS * 768 * 4);
  _Float16* hh   = (_Float16*)take((size_t)ROWS * 768 * 2);
  _Float16* gh   = (_Float16*)take((size_t)ROWS * 3072 * 2);

  auto cvtT = [&](const float* src, _Float16* dst, int K, int N) {
    cvt_transpose<<<dim3(N / 32, K / 32), 256, 0, stream>>>(src, dst, K, N);
  };
  cvtT(attn_w, wAt, 768, 2304);
  cvtT(proj_w, wPt, 768, 768);
  cvtT(fc_w,   wFt, 768, 3072);
  cvtT(fc2_w,  wF2t, 3072, 768);

  // LN1 -> f16
  ln_kernel<<<ROWS, 256, 0, stream>>>(x, ln1_w, ln1_b, xn, 768);
  // QKV: [4096,768] @ [768,2304] -> q,k (row layout) + v (transposed vT)
  gemm_kernel<<<dim3(2304 / 128, ROWS / 128), 256, 0, stream>>>(
      xn, wAt, attn_b, ROWS, 2304, 768, nullptr, qkvh, nullptr, 0, vT);
  // attention
  attn_kernel<<<2 * N_HEAD * (SEQ_T / 64), 128, 0, stream>>>(qkvh, vT, yh);
  // proj + residual(x) -> x2 (f32)
  gemm_kernel<<<dim3(768 / 128, ROWS / 128), 256, 0, stream>>>(
      yh, wPt, proj_b, ROWS, 768, 768, x2, nullptr, x, 0, nullptr);
  // LN2 -> f16
  ln_kernel<<<ROWS, 256, 0, stream>>>(x2, ln2_w, ln2_b, hh, 768);
  // FC + GELU -> f16
  gemm_kernel<<<dim3(3072 / 128, ROWS / 128), 256, 0, stream>>>(
      hh, wFt, fc_b, ROWS, 3072, 768, nullptr, gh, nullptr, 1, nullptr);
  // FC2 + residual(x2) -> d_out (f32)
  gemm_kernel<<<dim3(768 / 128, ROWS / 128), 256, 0, stream>>>(
      gh, wF2t, fc2_b, ROWS, 768, 3072, (float*)d_out, nullptr, x2, 0, nullptr);
}
